// EasyAttentionAggregator_42279658062562
// MI455X (gfx1250) — matrix operand order repaired and checked by reading, not compiled, not run on hardware
//
#include <hip/hip_runtime.h>
#include <stdint.h>

constexpr int kE = 128;     // embed dim
constexpr int kH = 4;       // heads
constexpr int kSeg = 128;   // number of segments

typedef __attribute__((ext_vector_type(16))) __bf16 v16bf;
typedef __attribute__((ext_vector_type(8)))  float  v8f;

union ABRegs { unsigned int u[8]; v16bf v; };

__device__ __forceinline__ unsigned int pack2(unsigned short lo, unsigned short hi) {
  return (unsigned int)lo | ((unsigned int)hi << 16);
}
__device__ __forceinline__ unsigned int packbf2(float a, float b) {
  unsigned short ua = __builtin_bit_cast(unsigned short, (__bf16)a);
  unsigned short ub = __builtin_bit_cast(unsigned short, (__bf16)b);
  return pack2(ua, ub);
}
__device__ __forceinline__ uint2 pack4(float a, float b, float c, float d) {
  uint2 p; p.x = packbf2(a, b); p.y = packbf2(c, d); return p;
}
// monotone float<->int order map (for integer atomicMax on floats)
__device__ __forceinline__ int f2ord(float f) { int i = __float_as_int(f); return i >= 0 ? i : (i ^ 0x7fffffff); }
__device__ __forceinline__ float ord2f(int i) { return __int_as_float(i >= 0 ? i : (i ^ 0x7fffffff)); }

// ---------------- segment boundary kernels ----------------
__global__ void seg_init(int* seg) {
  int t = threadIdx.x;
  if (t <= kSeg) seg[t] = 0x7fffffff;
}
__global__ void seg_bounds(const long long* __restrict__ batch, int* __restrict__ seg, int n) {
  int i = blockIdx.x * blockDim.x + threadIdx.x;
  if (i < n) atomicMin(&seg[(int)batch[i]], i);
}
__global__ void seg_fix(int* seg, int n) {
  if (blockIdx.x == 0 && threadIdx.x == 0) {
    seg[kSeg] = n;
    for (int b = kSeg - 1; b >= 0; --b)
      if (seg[b] > seg[b + 1]) seg[b] = seg[b + 1];   // empty segments collapse
  }
}

// ------- single-pass flash-style fused kernel: one workgroup per segment -------
__global__ __launch_bounds__(256)
void seg_attn(const float* __restrict__ x, const float* __restrict__ w,
              const int* __restrict__ seg, float* __restrict__ out) {
  __shared__ __align__(16) unsigned short xt[256 * kE]; // 64 KB bf16 stage (reused for merge)
  __shared__ unsigned short wt[16 * kE];                // 4 KB bf16 weights, zero-padded rows 4..15
  __shared__ int   mx[kH];
  __shared__ float ssum[kH];
  __shared__ float sinv[kH];

  const int b    = blockIdx.x;
  const int tid  = threadIdx.x;
  const int wave = tid >> 5;
  const int lane = tid & 31;
  const int col  = lane & 15;
  const int hi   = lane >> 4;
  const int ao   = hi ? 8 : 0;   // A-operand K sub-offset
  const int bo   = hi ? 16 : 0;  // B-operand K sub-offset

  const int s0 = seg[b], s1 = seg[b + 1];
  const int len = s1 - s0;
  if (len <= 0) {
    if (tid < kE) out[b * kE + tid] = 0.f;
    return;
  }

  for (int idx = tid; idx < 16 * kE; idx += 256)   // padded bf16 weights
    wt[idx] = (idx < kH * kE) ? (unsigned short)(packbf2(w[idx], 0.f) & 0xffffu)
                              : (unsigned short)0;
  if (tid < kH) { mx[tid] = f2ord(-3.0e38f); ssum[tid] = 0.f; }
  __syncthreads();

  const unsigned int* wt32 = (const unsigned int*)wt;
  const unsigned int* xt32 = (const unsigned int*)xt;

  // per-wave online-softmax state (value for head == col lives in lanes col, col+16)
  float mw = -3.0e38f;
  float sw = 0.f;
  v8f acc[8];
#pragma unroll
  for (int t = 0; t < 8; ++t) acc[t] = (v8f){0.f,0.f,0.f,0.f,0.f,0.f,0.f,0.f};

  const int nIt = (len + 255) >> 8;                // 256 rows per workgroup iteration
  for (int it = 0; it < nIt; ++it) {
    const int rowBase = s0 + (it << 8);
    // ---- stage 256 x-rows as bf16 (full / tail split) ----
    if (rowBase + 256 <= s1) {
      for (int i = 0; i < 32; ++i) {
        int idx = ((i << 8) + tid) << 2;           // float4 granularity
        int r = idx >> 7, c = idx & 127;
        const float4 v = *(const float4*)&x[(size_t)(rowBase + r) * kE + c];
        *(uint2*)&xt[r * kE + c] = pack4(v.x, v.y, v.z, v.w);
      }
    } else {
      for (int i = 0; i < 32; ++i) {
        int idx = ((i << 8) + tid) << 2;
        int r = idx >> 7, c = idx & 127;
        int gr = rowBase + r;
        float4 v = {0.f, 0.f, 0.f, 0.f};
        if (gr < s1) v = *(const float4*)&x[(size_t)gr * kE + c];
        *(uint2*)&xt[r * kE + c] = pack4(v.x, v.y, v.z, v.w);
      }
    }
    __syncthreads();

    const int tb = rowBase + (wave << 5);          // wave's 32-row chunk
    const int lr = wave << 5;

    // ---- GEMM 1: logits for 32 rows (two 16x16 tiles), K = E = 128 ----
    v8f c0 = (v8f){0.f,0.f,0.f,0.f,0.f,0.f,0.f,0.f};
    v8f c1 = (v8f){0.f,0.f,0.f,0.f,0.f,0.f,0.f,0.f};
#pragma unroll
    for (int j = 0; j < 4; ++j) {
      ABRegs a0, a1, bm;
      const int kb = j << 5;
#pragma unroll
      for (int p = 0; p < 4; ++p) {
        a0.u[p]     = xt32[(lr + col) * (kE / 2) + ((kb + ao + 2 * p) >> 1)];
        a0.u[4 + p] = xt32[(lr + col) * (kE / 2) + ((kb + 16 + ao + 2 * p) >> 1)];
        a1.u[p]     = xt32[(lr + 16 + col) * (kE / 2) + ((kb + ao + 2 * p) >> 1)];
        a1.u[4 + p] = xt32[(lr + 16 + col) * (kE / 2) + ((kb + 16 + ao + 2 * p) >> 1)];
      }
#pragma unroll
      for (int p = 0; p < 8; ++p)
        bm.u[p] = wt32[col * (kE / 2) + ((kb + ((p & 4) << 2) + ao + 2 * (p & 3)) >> 1)];
      c0 = __builtin_amdgcn_wmma_f32_16x16x32_bf16(false, a0.v, false, bm.v, (short)0, c0, false, false);
      c1 = __builtin_amdgcn_wmma_f32_16x16x32_bf16(false, a1.v, false, bm.v, (short)0, c1, false, false);
    }

    // ---- online softmax update (per head == col; exact up to offset invariance) ----
    float cm = c0[0];
#pragma unroll
    for (int r = 1; r < 8; ++r) cm = fmaxf(cm, c0[r]);
#pragma unroll
    for (int r = 0; r < 8; ++r) cm = fmaxf(cm, c1[r]);
    cm = fmaxf(cm, __shfl_xor(cm, 16, 32));
    const float mnew  = fmaxf(mw, cm);
    const float scale = __expf(mw - mnew);         // first iter: exp(-huge) -> 0
    mw = mnew;
#pragma unroll
    for (int r = 0; r < 8; ++r) { c0[r] = __expf(c0[r] - mnew); c1[r] = __expf(c1[r] - mnew); }

    float rs = 0.f;
    if (tb + 32 <= s1) {                           // full chunk: no masking
#pragma unroll
      for (int r = 0; r < 8; ++r) rs += c0[r] + c1[r];
    } else {                                       // tail: mask padded rows out of s
#pragma unroll
      for (int r = 0; r < 8; ++r) {
        if (tb + hi * 8 + r      < s1) rs += c0[r];
        if (tb + 16 + hi * 8 + r < s1) rs += c1[r];
      }
    }
    rs += __shfl_xor(rs, 16, 32);
    sw = sw * scale + rs;

    // rescale accumulators (head r lives in VGPR r; only heads 0..3 are read)
#pragma unroll
    for (int r = 0; r < kH; ++r) {
      const float sc = __shfl(scale, r, 32);
#pragma unroll
      for (int t = 0; t < 8; ++t) acc[t][r] *= sc;
    }

    // ---- GEMM 2: acc[head, e] += e^T x ; A comes straight from c0/c1 registers ----
    ABRegs ae;
#pragma unroll
    for (int p = 0; p < 4; ++p) {
      ae.u[p]     = packbf2(c0[2 * p], c0[2 * p + 1]);
      ae.u[4 + p] = packbf2(c1[2 * p], c1[2 * p + 1]);
    }
#pragma unroll
    for (int t = 0; t < 8; ++t) {
      ABRegs bm;
      const int ec = (t << 4) + col;
#pragma unroll
      for (int p = 0; p < 8; ++p) {
        const int k = lr + ((p & 4) << 2) + ao + 2 * (p & 3);
        bm.u[p] = pack2(xt[k * kE + ec], xt[(k + 1) * kE + ec]);
      }
      acc[t] = __builtin_amdgcn_wmma_f32_16x16x32_bf16(false, ae.v, false, bm.v, (short)0, acc[t], false, false);
    }
    __syncthreads();
  }

  // ---- cross-wave merge: M = max m_w ; s = sum s_w e^(m_w-M) ; h = sum acc_w e^(m_w-M) ----
  if (lane < kH) atomicMax(&mx[lane], f2ord(mw));
  __syncthreads();
  const float M  = ord2f(mx[lane & 3]);
  const float gs = __expf(mw - M);
  if (lane < kH) atomicAdd(&ssum[lane], sw * gs);

  float* red = (float*)xt;                         // reuse staging arena: 8 waves x 4 heads x 128
#pragma unroll
  for (int r = 0; r < kH; ++r) {
    const float sc = __shfl(gs, r, 32);
    if (lane < 16) {
#pragma unroll
      for (int t = 0; t < 8; ++t)
        red[((wave << 2) + r) * kE + (t << 4) + lane] = acc[t][r] * sc;
    }
  }
  __threadfence_block();
  __syncthreads();
  if (tid < kH) sinv[tid] = (ssum[tid] > 0.f) ? (1.f / ssum[tid]) : 0.f;
  __syncthreads();

  if (tid < kE) {                                  // out = mean over heads of h/s
    float o = 0.f;
#pragma unroll
    for (int r = 0; r < kH; ++r) {
      float hr = 0.f;
#pragma unroll
      for (int wv = 0; wv < 8; ++wv) hr += red[((wv << 2) + r) * kE + tid];
      o += hr * sinv[r];
    }
    out[b * kE + tid] = 0.25f * o;
  }
}

extern "C" void kernel_launch(void* const* d_in, const int* in_sizes, int n_in,
                              void* d_out, int out_size, void* d_ws, size_t ws_size,
                              hipStream_t stream) {
  const float*     x     = (const float*)d_in[0];
  const float*     w     = (const float*)d_in[1];
  const long long* batch = (const long long*)d_in[2];
  const int n = in_sizes[2];                       // N = 250000

  int* seg = (int*)d_ws;                           // (kSeg+1) ints

  seg_init  <<<1, 256, 0, stream>>>(seg);
  seg_bounds<<<(n + 255) / 256, 256, 0, stream>>>(batch, seg, n);
  seg_fix   <<<1, 1, 0, stream>>>(seg, n);
  seg_attn  <<<kSeg, 256, 0, stream>>>(x, w, seg, (float*)d_out);
}
